// Attention_90237262889748
// MI455X (gfx1250) — matrix operand order repaired and checked by reading, not compiled, not run on hardware
//
#include <hip/hip_runtime.h>
#include <hip/hip_bf16.h>
#include <math.h>

// ---------------- problem constants ----------------
#define BB   2
#define TT   2048
#define DD   2048
#define NH   16
#define NKV  4
#define HD   128
#define NREP (NH / NKV)            // 4
#define MM   (BB * TT)             // 4096 rows of "tokens"
#define NQ   (NH * HD * 2)         // 4096  (q + gate)
#define NKVC (NKV * HD)            // 512
#define DOUT (NH * HD)             // 2048
#define EPSF 1e-6f
#define SCALEF 0.08838834764831845f   // 128^-0.5

typedef float  v8f  __attribute__((ext_vector_type(8)));
typedef __bf16 v16bf __attribute__((ext_vector_type(16)));
typedef __bf16 v8bf  __attribute__((ext_vector_type(8)));

// ---------------- helpers ----------------
static __device__ __forceinline__ v8f zero_v8f() {
    v8f z;
#pragma unroll
    for (int i = 0; i < 8; ++i) z[i] = 0.0f;
    return z;
}

static __device__ __forceinline__ v8f wmma_bf16(v16bf a, v16bf b, v8f c) {
    // D = A(16x32 bf16) * B(32x16 bf16) + C(16x16 f32)
    return __builtin_amdgcn_wmma_f32_16x16x32_bf16(
        false, a, false, b, (short)0, c, false, false);
}

// A-fragment (16x32 bf16): lane L -> row = L&15, half = L>>4.
// element e -> K = half*16 + e
// src is row-major [rows][ld] bf16.
static __device__ __forceinline__ v16bf load_a_frag(const __bf16* src, int ld,
                                                    int row0, int kbase) {
    int lane = threadIdx.x & 31;
    int half = lane >> 4;
    const __bf16* p = src + (size_t)(row0 + (lane & 15)) * ld + kbase + half * 16;
    v8bf lo = *(const v8bf*)(p);
    v8bf hi = *(const v8bf*)(p + 8);
    v16bf a;
#pragma unroll
    for (int i = 0; i < 8; ++i) { a[i] = lo[i]; a[i + 8] = hi[i]; }
    return a;
}

// B-fragment (32x16 bf16): lane L -> col = L&15, half = L>>4, element e -> K = half*16 + e.
// srcT is B transposed, row-major [Ncols][ld] bf16, so the 16 K-values are contiguous.
static __device__ __forceinline__ v16bf load_b_frag(const __bf16* srcT, int ld,
                                                    int col0, int kbase) {
    int lane = threadIdx.x & 31;
    int half = lane >> 4;
    return *(const v16bf*)(srcT + (size_t)(col0 + (lane & 15)) * ld + kbase + half * 16);
}

// per-wave LDS ordering fence (DS ops are in-order within a wave; stop compiler motion)
static __device__ __forceinline__ void wave_sync() {
    __builtin_amdgcn_fence(__ATOMIC_RELEASE, "wavefront");
    __builtin_amdgcn_wave_barrier();
    __builtin_amdgcn_fence(__ATOMIC_ACQUIRE, "wavefront");
}

// ---------------- conversion kernels ----------------
__global__ void cvt_f32_to_bf16(const float* __restrict__ in,
                                __bf16* __restrict__ out, int n) {
    int i = blockIdx.x * blockDim.x + threadIdx.x;
    if (i < n) out[i] = (__bf16)in[i];
}

// in: [K][N] f32 (row-major), out: [N][K] bf16 (transposed)
__global__ void transpose_cvt_bf16(const float* __restrict__ in,
                                   __bf16* __restrict__ out, int K, int N) {
    int i = blockIdx.x * blockDim.x + threadIdx.x;
    if (i < K * N) {
        int k = i / N, n = i - k * N;
        out[(size_t)n * K + k] = (__bf16)in[i];
    }
}

// ---------------- WMMA GEMM with async-DMA LDS double buffering ----------------
// C[M][N] f32 = A[M][K] bf16 * B[K][N] bf16 (B given transposed, [N][K]).
// Workgroup: 256 threads = 8 waves, computes a 128x128 C tile.
// Per 32-wide K step: A panel 128x32 bf16 (8KB) + Bt panel 128x32 bf16 (8KB) are
// DMA'd into LDS with GLOBAL_LOAD_ASYNC_TO_LDS_B128 (ASYNCcnt), double buffered.
// Waves 0..3 move the A panel, waves 4..7 the Bt panel; all 8 waves compute 32x64.
__global__ void gemm_bf16_wmma_async(const __bf16* __restrict__ A,
                                     const __bf16* __restrict__ Bt,
                                     float* __restrict__ C, int M, int N, int K) {
    // [buf0: A 8KB | Bt 8KB][buf1: A 8KB | Bt 8KB]
    __shared__ __align__(32) char smem[2 * 16384];

    int lane = threadIdx.x & 31;
    int wid  = threadIdx.x >> 5;
    int tiles_n = N >> 7;
    int mt = blockIdx.x / tiles_n;
    int nt = blockIdx.x - mt * tiles_n;
    int row0 = mt << 7;          // first A row of this block
    int col0 = nt << 7;          // first B column (= row of Bt)

    // DMA role of this wave
    int grp = wid & 3;           // which quarter of the 8KB panel
    int isB = wid >> 2;          // 0 = A panel, 1 = Bt panel
    const char* gsrc = isB ? (const char*)(Bt + (size_t)col0 * K)
                           : (const char*)(A  + (size_t)row0 * K);
    size_t rowstride = (size_t)K * 2;    // bytes per panel row (row of A / row of Bt)

    // Each wave issues 4 async b128 ops = 2KB; 4 waves per panel = 8KB.
    // LDS address operand is derived from the shared-array pointer itself:
    //  - low 32 bits of a flat LDS address are the workgroup-relative LDS offset
    //  - ptrtoint captures smem so the "memory" clobber is honored and the
    //    subsequent LDS fragment reads are not folded away.
    auto issue = [&](int buf, int kb) {
#pragma unroll
        for (int i = 0; i < 4; ++i) {
            int byteoff = grp * 2048 + i * 512 + lane * 16;   // within 8KB panel
            int prow = byteoff >> 6;                          // 64B per panel row
            int kB   = byteoff & 63;                          // byte within row
            char* lp = smem + buf * 16384 + isB * 8192 + byteoff;
            unsigned lds_off = (unsigned)(uintptr_t)lp;
            unsigned long long gaddr = (unsigned long long)(uintptr_t)
                (gsrc + (size_t)prow * rowstride + (size_t)kb * 2 + kB);
            asm volatile("global_load_async_to_lds_b128 %0, %1, off"
                         :: "v"(lds_off), "v"(gaddr) : "memory");
        }
    };

    // compute sub-tile of this wave: 32 rows x 64 cols
    int moff = (wid >> 1) * 32;
    int noff = (wid & 1) * 64;

    v8f acc[2][4];
#pragma unroll
    for (int mi = 0; mi < 2; ++mi)
#pragma unroll
        for (int t = 0; t < 4; ++t) acc[mi][t] = zero_v8f();

    issue(0, 0);                 // prologue fill of buffer 0
    int buf = 0;
    for (int kb = 0; kb < K; kb += 32) {
        if (kb + 32 < K) {
            issue(buf ^ 1, kb + 32);                     // prefetch next panel
            asm volatile("s_wait_asynccnt 4" ::: "memory");   // current panel done
        } else {
            asm volatile("s_wait_asynccnt 0" ::: "memory");
        }
        __syncthreads();                                 // publish LDS across waves

        const __bf16* Al = (const __bf16*)(smem + buf * 16384);
        const __bf16* Bl = (const __bf16*)(smem + buf * 16384 + 8192);
        v16bf af0 = load_a_frag(Al, 32, moff, 0);
        v16bf af1 = load_a_frag(Al, 32, moff + 16, 0);
#pragma unroll
        for (int t = 0; t < 4; ++t) {
            v16bf bf = load_b_frag(Bl, 32, noff + t * 16, 0);
            acc[0][t] = wmma_bf16(af0, bf, acc[0][t]);
            acc[1][t] = wmma_bf16(af1, bf, acc[1][t]);
        }
        __syncthreads();                                 // done reading before overwrite
        buf ^= 1;
    }

    int n = lane & 15, half = lane >> 4;
#pragma unroll
    for (int mi = 0; mi < 2; ++mi)
#pragma unroll
        for (int t = 0; t < 4; ++t)
#pragma unroll
            for (int r = 0; r < 8; ++r)
                C[(size_t)(row0 + moff + mi * 16 + r + half * 8) * N
                  + col0 + noff + t * 16 + n] = acc[mi][t][r];
}

// ---------------- RMSNorm + RoPE epilogues ----------------
// q_raw: [MM][NQ] f32.  Writes q_bf16 [B][NH][T][HD] and gate f32 [MM][NH*HD].
__global__ void post_q_kernel(const float* __restrict__ q_raw,
                              const float* __restrict__ cosb,
                              const float* __restrict__ sinb,
                              const float* __restrict__ qnw,
                              __bf16* __restrict__ q_out,
                              float* __restrict__ gate_out) {
    int m = blockIdx.x / NH;
    int h = blockIdx.x - m * NH;
    int hd = threadIdx.x;
    int b = m / TT, tpos = m - b * TT;

    __shared__ float red[HD];
    __shared__ float nv[HD];

    float val = q_raw[(size_t)m * NQ + h * (2 * HD) + hd];
    float gv  = q_raw[(size_t)m * NQ + h * (2 * HD) + HD + hd];

    red[hd] = val * val;
    __syncthreads();
    for (int s = HD / 2; s > 0; s >>= 1) {
        if (hd < s) red[hd] += red[hd + s];
        __syncthreads();
    }
    float rms = rsqrtf(red[0] * (1.0f / HD) + EPSF);
    float normed = val * rms * qnw[hd];
    nv[hd] = normed;
    __syncthreads();
    float rot = (hd < HD / 2) ? -nv[hd + HD / 2] : nv[hd - HD / 2];
    size_t ci = (size_t)m * HD + hd;
    float o = normed * cosb[ci] + rot * sinb[ci];
    q_out[((size_t)(b * NH + h) * TT + tpos) * HD + hd] = (__bf16)o;
    gate_out[(size_t)m * DOUT + h * HD + hd] = gv;
}

// k_raw/v_raw: [MM][NKVC] f32. Writes k_bf16 [B][NKV][T][HD] and vT bf16 [B][NKV][HD][T].
__global__ void post_kv_kernel(const float* __restrict__ k_raw,
                               const float* __restrict__ v_raw,
                               const float* __restrict__ cosb,
                               const float* __restrict__ sinb,
                               const float* __restrict__ knw,
                               __bf16* __restrict__ k_out,
                               __bf16* __restrict__ vt_out) {
    int m = blockIdx.x / NKV;
    int g = blockIdx.x - m * NKV;
    int hd = threadIdx.x;
    int b = m / TT, tpos = m - b * TT;

    __shared__ float red[HD];
    __shared__ float nv[HD];

    float val = k_raw[(size_t)m * NKVC + g * HD + hd];
    red[hd] = val * val;
    __syncthreads();
    for (int s = HD / 2; s > 0; s >>= 1) {
        if (hd < s) red[hd] += red[hd + s];
        __syncthreads();
    }
    float rms = rsqrtf(red[0] * (1.0f / HD) + EPSF);
    float normed = val * rms * knw[hd];
    nv[hd] = normed;
    __syncthreads();
    float rot = (hd < HD / 2) ? -nv[hd + HD / 2] : nv[hd - HD / 2];
    size_t ci = (size_t)m * HD + hd;
    float o = normed * cosb[ci] + rot * sinb[ci];
    k_out[((size_t)(b * NKV + g) * TT + tpos) * HD + hd] = (__bf16)o;

    float vv = v_raw[(size_t)m * NKVC + g * HD + hd];
    vt_out[((size_t)(b * NKV + g) * HD + hd) * TT + tpos] = (__bf16)vv;
}

// ---------------- flash attention (one wave = 16 query rows) ----------------
// q: [B][NH][T][HD] bf16 ; k: [B][NKV][T][HD] bf16 ; vt: [B][NKV][HD][T] bf16
// gate: [MM][NH*HD] f32 ; out: [MM][NH*HD] bf16 (gated attention output)
__global__ void flash_attn_wmma(const __bf16* __restrict__ q,
                                const __bf16* __restrict__ k,
                                const __bf16* __restrict__ vt,
                                const float* __restrict__ gate,
                                __bf16* __restrict__ out) {
    __shared__ float  S_lds[4][16 * 32];
    __shared__ __align__(16) __bf16 P_lds[4][16 * 32];

    int lane = threadIdx.x & 31;
    int wid  = threadIdx.x >> 5;
    int task = blockIdx.x * 4 + wid;
    int total = BB * NH * (TT / 16);
    if (task >= total) return;

    int qt = task & 127;                 // T/16 = 128 query tiles
    int h  = (task >> 7) & (NH - 1);
    int b  = task >> 11;
    int kvh = h >> 2;                    // NREP = 4
    int half = lane >> 4;
    int n = lane & 15;

    const __bf16* qbase = q  + ((size_t)(b * NH  + h)   * TT) * HD;
    const __bf16* kbase = k  + ((size_t)(b * NKV + kvh) * TT) * HD;
    const __bf16* vbase = vt + ((size_t)(b * NKV + kvh) * HD) * TT;

    // Q fragments for the whole HD=128 (4 chunks of K=32)
    v16bf qf[4];
#pragma unroll
    for (int kk = 0; kk < 4; ++kk)
        qf[kk] = load_a_frag(qbase, HD, qt * 16, kk * 32);

    v8f o[8];
#pragma unroll
    for (int t = 0; t < 8; ++t) o[t] = zero_v8f();

    float m_run = -1e30f, l_run = 0.0f;   // valid on lanes 0..15

    int kmaxrow = qt * 16 + 15;
    int nsteps = (kmaxrow + 32) / 32;     // ceil((qt*16+16)/32)

    for (int s = 0; s < nsteps; ++s) {
        int kb = s * 32;
        // ---- S = Q * K^T (two 16-key column tiles) ----
        v8f s0 = zero_v8f(), s1 = zero_v8f();
#pragma unroll
        for (int kk = 0; kk < 4; ++kk) {
            v16bf b0 = load_b_frag(kbase, HD, kb,      kk * 32);
            v16bf b1 = load_b_frag(kbase, HD, kb + 16, kk * 32);
            s0 = wmma_bf16(qf[kk], b0, s0);
            s1 = wmma_bf16(qf[kk], b1, s1);
        }
        // store scaled S tile (16x32) into this wave's LDS region
        float* Sl = S_lds[wid];
#pragma unroll
        for (int r = 0; r < 8; ++r) {
            int row = r + half * 8;
            Sl[row * 32 + n]      = s0[r] * SCALEF;
            Sl[row * 32 + 16 + n] = s1[r] * SCALEF;
        }
        wave_sync();

        // ---- online softmax: lanes 0..15 each own one query row ----
        float corr = 1.0f;
        if (lane < 16) {
            int row = lane;
            int q_abs = qt * 16 + row;
            const float* Sr = &S_lds[wid][row * 32];
            float mx = -1e30f;
#pragma unroll
            for (int j = 0; j < 32; ++j) {
                float x = (kb + j <= q_abs) ? Sr[j] : -1e30f;
                mx = fmaxf(mx, x);
            }
            float m_new = fmaxf(m_run, mx);
            corr = __expf(m_run - m_new);
            float rs = 0.0f;
            __bf16* Pr = &P_lds[wid][row * 32];
#pragma unroll
            for (int j = 0; j < 32; ++j) {
                float x = (kb + j <= q_abs) ? Sr[j] : -1e30f;
                float p = __expf(x - m_new);
                rs += p;
                Pr[j] = (__bf16)p;
            }
            l_run = l_run * corr + rs;
            m_run = m_new;
        }
        wave_sync();

        // ---- rescale O accumulators by exp(m_old - m_new), broadcast per row ----
#pragma unroll
        for (int r = 0; r < 8; ++r) {
            float sc = __shfl(corr, half ? (r + 8) : r, 32);
#pragma unroll
            for (int t = 0; t < 8; ++t) o[t][r] *= sc;
        }

        // ---- O += P * V  (P 16x32 from LDS as A-frag, V^T contiguous as B-frag) ----
        v16bf pf = load_a_frag(P_lds[wid], 32, 0, 0);
#pragma unroll
        for (int t = 0; t < 8; ++t) {
            v16bf vf = load_b_frag(vbase, TT, t * 16, kb);
            o[t] = wmma_bf16(pf, vf, o[t]);
        }
        wave_sync();
    }

    // ---- epilogue: divide by row-sum, gate with sigmoid, store bf16 ----
    float linv_src = (lane < 16) ? (1.0f / l_run) : 0.0f;
#pragma unroll
    for (int r = 0; r < 8; ++r) {
        float li = __shfl(linv_src, half ? (r + 8) : r, 32);
        size_t rowoff = (size_t)(b * TT + qt * 16 + r + half * 8) * DOUT + h * HD + n;
#pragma unroll
        for (int t = 0; t < 8; ++t) {
            float val = o[t][r] * li;
            float g = gate[rowoff + t * 16];
            val *= 1.0f / (1.0f + __expf(-g));
            out[rowoff + t * 16] = (__bf16)val;
        }
    }
}

// ---------------- host-side orchestration ----------------
extern "C" void kernel_launch(void* const* d_in, const int* in_sizes, int n_in,
                              void* d_out, int out_size, void* d_ws, size_t ws_size,
                              hipStream_t stream) {
    const float* hidden = (const float*)d_in[0];   // [B,T,D]
    const float* cosb   = (const float*)d_in[1];   // [B,T,HD]
    const float* sinb   = (const float*)d_in[2];   // [B,T,HD]
    // d_in[3] segment_ids, d_in[4] position_ids: unused (trivial in reference)
    const float* w_q    = (const float*)d_in[5];   // [D, NQ]
    const float* w_k    = (const float*)d_in[6];   // [D, NKVC]
    const float* w_v    = (const float*)d_in[7];   // [D, NKVC]
    const float* w_o    = (const float*)d_in[8];   // [DOUT, D]
    const float* qnw    = (const float*)d_in[9];   // [HD]
    const float* knw    = (const float*)d_in[10];  // [HD]
    float* out_f32 = (float*)d_out;                // [B,T,D]

    char* ws = (char*)d_ws;
    size_t off = 0;
    auto alloc = [&](size_t bytes) {
        size_t r = off;
        off += (bytes + 255) & ~(size_t)255;
        return r;
    };

    __bf16* hiddenB = (__bf16*)(ws + alloc((size_t)MM * DD * 2));            // [MM][DD]
    __bf16* wqT     = (__bf16*)(ws + alloc((size_t)NQ * DD * 2));            // [NQ][DD]
    __bf16* wkT     = (__bf16*)(ws + alloc((size_t)NKVC * DD * 2));          // [NKVC][DD]
    __bf16* wvT     = (__bf16*)(ws + alloc((size_t)NKVC * DD * 2));          // [NKVC][DD]
    __bf16* woT     = (__bf16*)(ws + alloc((size_t)DD * DOUT * 2));          // [DD][DOUT]
    float*  q_raw   = (float*)(ws + alloc((size_t)MM * NQ * 4));             // [MM][NQ]
    float*  k_raw   = (float*)(ws + alloc((size_t)MM * NKVC * 4));           // [MM][NKVC]
    float*  v_raw   = (float*)(ws + alloc((size_t)MM * NKVC * 4));           // [MM][NKVC]
    __bf16* q_bf    = (__bf16*)(ws + alloc((size_t)BB * NH * TT * HD * 2));  // [B][NH][T][HD]
    __bf16* k_bf    = (__bf16*)(ws + alloc((size_t)BB * NKV * TT * HD * 2)); // [B][NKV][T][HD]
    __bf16* v_t     = (__bf16*)(ws + alloc((size_t)BB * NKV * HD * TT * 2)); // [B][NKV][HD][T]
    float*  gate_f  = (float*)(ws + alloc((size_t)MM * DOUT * 4));           // [MM][DOUT]
    __bf16* attn_g  = (__bf16*)(ws + alloc((size_t)MM * DOUT * 2));          // [MM][DOUT]
    (void)ws_size; (void)in_sizes; (void)n_in; (void)out_size;

    // 1) convert hidden to bf16
    {
        int nelem = MM * DD;
        cvt_f32_to_bf16<<<(nelem + 255) / 256, 256, 0, stream>>>(hidden, hiddenB, nelem);
    }
    // 2) transpose+convert weights to bf16 [N][K]
    transpose_cvt_bf16<<<(DD * NQ + 255) / 256, 256, 0, stream>>>(w_q, wqT, DD, NQ);
    transpose_cvt_bf16<<<(DD * NKVC + 255) / 256, 256, 0, stream>>>(w_k, wkT, DD, NKVC);
    transpose_cvt_bf16<<<(DD * NKVC + 255) / 256, 256, 0, stream>>>(w_v, wvT, DD, NKVC);
    transpose_cvt_bf16<<<(DOUT * DD + 255) / 256, 256, 0, stream>>>(w_o, woT, DOUT, DD);

    // 3) projections (async-DMA double-buffered WMMA GEMMs), f32 outputs
    gemm_bf16_wmma_async<<<(MM / 128) * (NQ / 128), 256, 0, stream>>>(
        hiddenB, wqT, q_raw, MM, NQ, DD);
    gemm_bf16_wmma_async<<<(MM / 128) * (NKVC / 128), 256, 0, stream>>>(
        hiddenB, wkT, k_raw, MM, NKVC, DD);
    gemm_bf16_wmma_async<<<(MM / 128) * (NKVC / 128), 256, 0, stream>>>(
        hiddenB, wvT, v_raw, MM, NKVC, DD);

    // 4) RMSNorm + RoPE epilogues, layout shuffles
    post_q_kernel<<<MM * NH, HD, 0, stream>>>(q_raw, cosb, sinb, qnw, q_bf, gate_f);
    post_kv_kernel<<<MM * NKV, HD, 0, stream>>>(k_raw, v_raw, cosb, sinb, knw, k_bf, v_t);

    // 5) flash attention + gating (one wave per 16 query rows)
    {
        int tasks = BB * NH * (TT / 16);      // 4096 waves
        flash_attn_wmma<<<tasks / 4, 128, 0, stream>>>(q_bf, k_bf, v_t, gate_f, attn_g);
    }

    // 6) output projection straight into d_out (f32)
    gemm_bf16_wmma_async<<<(MM / 128) * (DD / 128), 256, 0, stream>>>(
        attn_g, woT, out_f32, MM, DD, DOUT);
}
